// QWEN4TS_70609262346546
// MI455X (gfx1250) — matrix operand order repaired and checked by reading, not compiled, not run on hardware
//
#include <hip/hip_runtime.h>
#include <hip/hip_bf16.h>
#include <math.h>

typedef __bf16 bf16;
typedef __attribute__((ext_vector_type(16))) __bf16 v16bf;
typedef __attribute__((ext_vector_type(8)))  __bf16 v8bf;
typedef __attribute__((ext_vector_type(8)))  float  v8f;

#define D_MODEL 3584
#define N_HEADS 28
#define N_KV    4
#define HEAD_DIM 128
#define D_FF    18944
#define SEQ     64
#define NSER    28
#define NTOK    1792
#define PRED    96
#define OUTK    (D_MODEL * SEQ)   // 229376

__device__ __forceinline__ float bf2f(bf16 v) { return (float)v; }
__device__ __forceinline__ bf16  f2bf(float v) { return (bf16)v; }

__device__ __forceinline__ v8f v8f_zero() {
  v8f z;
#pragma unroll
  for (int i = 0; i < 8; ++i) z[i] = 0.0f;
  return z;
}

__device__ __forceinline__ v8f wmma_bf16(v16bf a, v16bf b, v8f c) {
  return __builtin_amdgcn_wmma_f32_16x16x32_bf16(false, a, false, b, (short)0, c, false, false);
}

// Async global->LDS copy of 16 bytes (per lane), tracked by ASYNCcnt.
__device__ __forceinline__ void async_copy_b128(unsigned lds_addr, const void* gaddr) {
  asm volatile("global_load_async_to_lds_b128 %0, %1, off"
               :: "v"(lds_addr), "v"(gaddr) : "memory");
}
__device__ __forceinline__ void wait_async0() {
#if __has_builtin(__builtin_amdgcn_s_wait_asynccnt)
  __builtin_amdgcn_s_wait_asynccnt(0);
#else
  asm volatile("s_wait_asynccnt 0x0" ::: "memory");
#endif
}

// A-fragment (16x32, M=lane%16). Lane<16: K 0-7 then 16-23; lane>=16: K 8-15 then 24-31.
__device__ __forceinline__ v16bf load_a_frag(const bf16* row, int lane) {
  const int off = (lane < 16) ? 0 : 8;
  v8bf lo = *(const v8bf*)(row + off);
  v8bf hi = *(const v8bf*)(row + off + 16);
  v16bf r;
#pragma unroll
  for (int i = 0; i < 8; ++i) { r[i] = lo[i]; r[i + 8] = hi[i]; }
  return r;
}

// B-fragment from transposed storage Bt[n][k] (contiguous K). Lane<16: K 0-7 then 16-23; lane>=16: K 8-15 then 24-31.
__device__ __forceinline__ v16bf load_b_frag(const bf16* rowT, int lane) {
  const int off = (lane < 16) ? 0 : 8;
  v8bf lo = *(const v8bf*)(rowT + off);
  v8bf hi = *(const v8bf*)(rowT + off + 16);
  v16bf r;
#pragma unroll
  for (int i = 0; i < 8; ++i) { r[i] = lo[i]; r[i + 8] = hi[i]; }
  return r;
}

// ---------------- RevIN stats ----------------
__global__ void stats_kernel(const float* __restrict__ x, float* __restrict__ stats) {
  const int ser = blockIdx.x;                // b*7+m
  const int b = ser / 7, m = ser % 7;
  __shared__ float s1[256], s2[256];
  float a = 0.f, q = 0.f;
  for (int t = threadIdx.x; t < 512; t += 256) {
    float v = x[((size_t)b * 512 + t) * 7 + m];
    a += v; q += v * v;
  }
  s1[threadIdx.x] = a; s2[threadIdx.x] = q;
  __syncthreads();
  for (int s = 128; s > 0; s >>= 1) {
    if (threadIdx.x < s) { s1[threadIdx.x] += s1[threadIdx.x + s]; s2[threadIdx.x] += s2[threadIdx.x + s]; }
    __syncthreads();
  }
  if (threadIdx.x == 0) {
    float mean = s1[0] / 512.f;
    float var  = s2[0] / 512.f - mean * mean;
    stats[ser * 2]     = mean;
    stats[ser * 2 + 1] = sqrtf(var + 1e-5f);
  }
}

// ---------------- patch embed ----------------
__global__ void embed_kernel(const float* __restrict__ x, const float* __restrict__ stats,
                             const float* __restrict__ in_w, const float* __restrict__ in_b,
                             bf16* __restrict__ h) {
  const int tok = blockIdx.x;          // ser*64 + patch
  const int ser = tok >> 6;
  const int p   = tok & 63;
  const int b = ser / 7, m = ser % 7;
  __shared__ float pv[16];
  if (threadIdx.x < 16) {
    int t = p * 8 + threadIdx.x;
    if (t > 511) t = 511;              // edge padding
    float mean = stats[ser * 2], sd = stats[ser * 2 + 1];
    pv[threadIdx.x] = (x[((size_t)b * 512 + t) * 7 + m] - mean) / sd;
  }
  __syncthreads();
  for (int d = threadIdx.x; d < D_MODEL; d += 256) {
    float acc = in_b[d];
#pragma unroll
    for (int j = 0; j < 16; ++j) acc += pv[j] * in_w[j * D_MODEL + d];
    h[(size_t)tok * D_MODEL + d] = f2bf(acc);
  }
}

// ---------------- RMSNorm ----------------
__global__ void rmsnorm_kernel(const bf16* __restrict__ h, const bf16* __restrict__ w,
                               bf16* __restrict__ o) {
  const int tok = blockIdx.x;
  __shared__ float red[256];
  const bf16* row = h + (size_t)tok * D_MODEL;
  float q = 0.f;
  for (int d = threadIdx.x; d < D_MODEL; d += 256) { float v = bf2f(row[d]); q += v * v; }
  red[threadIdx.x] = q; __syncthreads();
  for (int s = 128; s > 0; s >>= 1) {
    if (threadIdx.x < s) red[threadIdx.x] += red[threadIdx.x + s];
    __syncthreads();
  }
  const float r = rsqrtf(red[0] / (float)D_MODEL + 1e-6f);
  bf16* orow = o + (size_t)tok * D_MODEL;
  for (int d = threadIdx.x; d < D_MODEL; d += 256)
    orow[d] = f2bf(bf2f(row[d]) * r * bf2f(w[d]));
}

// ---------------- RoPE (in-place, position = patch index within series) ----------------
__global__ void rope_kernel(bf16* __restrict__ qk, int nheads, int ntok) {
  const int total = ntok * nheads * 64;
  int i = blockIdx.x * 256 + threadIdx.x;
  if (i >= total) return;
  const int d  = i & 63;
  int t = i >> 6;
  const int hh  = t % nheads;
  const int tok = t / nheads;
  const int pos = tok & 63;
  float inv = powf(1.0e6f, -(float)d / 64.f);
  float ang = (float)pos * inv;
  float c = cosf(ang), s = sinf(ang);
  bf16* base = qk + (size_t)tok * (nheads * HEAD_DIM) + hh * HEAD_DIM;
  float x0 = bf2f(base[d]), x1 = bf2f(base[d + 64]);
  base[d]      = f2bf(x0 * c - x1 * s);
  base[d + 64] = f2bf(x1 * c + x0 * s);
}

// ---------------- WMMA GEMM: C[M,N] = A[M,K] @ W[K,N] (+epilogue) ----------------
// Double-buffered LDS; A tile streamed with GLOBAL_LOAD_ASYNC_TO_LDS_B128 (ASYNCcnt),
// W tile register-staged and stored transposed so B-fragments are contiguous ds_load_b128.
// EPI 0: C = bf16(acc + bias)          (bias optional)
// EPI 1: C = bf16(float(C) + acc)      (residual accumulate)
// EPI 2: C = bf16(silu(float(C))*acc)  (C holds gate; up-proj fuse)
template <int EPI>
__global__ __launch_bounds__(256)
void gemm_bf16(const bf16* __restrict__ A, const bf16* __restrict__ W,
               bf16* __restrict__ C, const bf16* __restrict__ bias,
               int M, int N, int K) {
  __shared__ bf16 As[2][128][40];
  __shared__ bf16 Bt[2][128][40];
  const int tid  = threadIdx.x;
  const int lane = tid & 31;
  const int wave = tid >> 5;
  const int wm = wave >> 2;          // 0..1 -> 64-row half
  const int wn = wave & 3;           // 0..3 -> 32-col quarter
  const int m0 = blockIdx.y * 128;
  const int n0 = blockIdx.x * 128;

  v8f acc[4][2];
#pragma unroll
  for (int i = 0; i < 4; ++i)
#pragma unroll
    for (int j = 0; j < 2; ++j) acc[i][j] = v8f_zero();

  const int ar = tid >> 1, asel = (tid & 1) * 16;   // A: 2 threads/row, 16 halves each
  const int bkr = tid >> 3, bnc = (tid & 7) * 16;   // W: 8 threads/row, 16 halves each

  const bf16* arow_base = A + (size_t)(m0 + ar) * K + asel;
  v8bf wb0, wb1;

  const int ktiles = K >> 5;

  // ---- prologue: tile 0 into stage 0 ----
  {
    const bf16* ga = arow_base;                                  // k0 = 0
    async_copy_b128((unsigned)(uintptr_t)&As[0][ar][asel],     ga);
    async_copy_b128((unsigned)(uintptr_t)&As[0][ar][asel + 8], ga + 8);
    const bf16* gw = W + (size_t)bkr * N + n0 + bnc;
    wb0 = *(const v8bf*)gw;
    wb1 = *(const v8bf*)(gw + 8);
#pragma unroll
    for (int i = 0; i < 8; ++i) Bt[0][bnc + i][bkr] = wb0[i];
#pragma unroll
    for (int i = 0; i < 8; ++i) Bt[0][bnc + 8 + i][bkr] = wb1[i];
  }
  wait_async0();
  __syncthreads();

  for (int kt = 0; kt < ktiles; ++kt) {
    const int stage = kt & 1;
    const int next  = stage ^ 1;
    const bool more = (kt + 1) < ktiles;

    if (more) {
      const int k1 = (kt + 1) << 5;
      // next A tile: async straight into LDS
      const bf16* ga = arow_base + k1;
      async_copy_b128((unsigned)(uintptr_t)&As[next][ar][asel],     ga);
      async_copy_b128((unsigned)(uintptr_t)&As[next][ar][asel + 8], ga + 8);
      // next W tile: into registers (scattered after compute)
      const bf16* gw = W + (size_t)(k1 + bkr) * N + n0 + bnc;
      wb0 = *(const v8bf*)gw;
      wb1 = *(const v8bf*)(gw + 8);
      __builtin_prefetch(gw + (size_t)32 * N, 0, 0);
    }

    // ---- compute on current stage ----
    v16bf afr[4];
#pragma unroll
    for (int i = 0; i < 4; ++i)
      afr[i] = load_a_frag(&As[stage][wm * 64 + i * 16 + (lane & 15)][0], lane);
#pragma unroll
    for (int j = 0; j < 2; ++j) {
      v16bf bfr = load_b_frag(&Bt[stage][wn * 32 + j * 16 + (lane & 15)][0], lane);
#pragma unroll
      for (int i = 0; i < 4; ++i)
        acc[i][j] = wmma_bf16(afr[i], bfr, acc[i][j]);
    }

    if (more) {
#pragma unroll
      for (int i = 0; i < 8; ++i) Bt[next][bnc + i][bkr] = wb0[i];
#pragma unroll
      for (int i = 0; i < 8; ++i) Bt[next][bnc + 8 + i][bkr] = wb1[i];
    }
    wait_async0();
    __syncthreads();
  }

#pragma unroll
  for (int i = 0; i < 4; ++i) {
#pragma unroll
    for (int j = 0; j < 2; ++j) {
#pragma unroll
      for (int r = 0; r < 8; ++r) {
        const int row = m0 + wm * 64 + i * 16 + ((lane < 16) ? r : r + 8);
        const int col = n0 + wn * 32 + j * 16 + (lane & 15);
        const size_t idx = (size_t)row * N + col;
        float v = acc[i][j][r];
        if (EPI == 0) {
          if (bias) v += bf2f(bias[col]);
          C[idx] = f2bf(v);
        } else if (EPI == 1) {
          C[idx] = f2bf(bf2f(C[idx]) + v);
        } else {
          float g = bf2f(C[idx]);
          C[idx] = f2bf((g / (1.f + expf(-g))) * v);
        }
      }
    }
  }
}

// ---------------- fused GQA attention: one block per (series, kv-head) ----------------
__global__ __launch_bounds__(256)
void attn_kernel(const bf16* __restrict__ q, const bf16* __restrict__ k,
                 const bf16* __restrict__ v, bf16* __restrict__ o) {
  const int ser  = blockIdx.x >> 2;
  const int kvh  = blockIdx.x & 3;
  const int tok0 = ser * SEQ;
  const int tid = threadIdx.x, lane = tid & 31, wave = tid >> 5;

  __shared__ bf16  vt[HEAD_DIM][SEQ];   // V transposed: vt[d][ki]
  __shared__ float sc[SEQ][68];         // fp32 scores
  __shared__ bf16  pb[SEQ][72];         // bf16 probabilities

  for (int idx = tid; idx < SEQ * HEAD_DIM; idx += 256) {
    int ki = idx >> 7, d = idx & 127;
    vt[d][ki] = v[(size_t)(tok0 + ki) * (N_KV * HEAD_DIM) + kvh * HEAD_DIM + d];
  }
  __syncthreads();

  const float scale = 0.08838834764831845f; // 1/sqrt(128)
  for (int qh = 0; qh < N_HEADS / N_KV; ++qh) {
    const int head = kvh * (N_HEADS / N_KV) + qh;

    // scores = Q K^T * scale, causal mask -> sc
    const int sm  = wave >> 1;           // q-row subtile 0..3
    const int sn0 = (wave & 1) * 2;      // 2 key subtiles
    v8f accs[2] = { v8f_zero(), v8f_zero() };
    const int qrow = sm * 16 + (lane & 15);
    const bf16* qbase = q + (size_t)(tok0 + qrow) * (N_HEADS * HEAD_DIM) + head * HEAD_DIM;
#pragma unroll
    for (int d0 = 0; d0 < HEAD_DIM; d0 += 32) {
      v16bf af = load_a_frag(qbase + d0, lane);
#pragma unroll
      for (int j = 0; j < 2; ++j) {
        const int kcol = (sn0 + j) * 16 + (lane & 15);
        const bf16* kbase = k + (size_t)(tok0 + kcol) * (N_KV * HEAD_DIM) + kvh * HEAD_DIM + d0;
        v16bf bfrag = load_b_frag(kbase, lane);
        accs[j] = wmma_bf16(af, bfrag, accs[j]);
      }
    }
#pragma unroll
    for (int j = 0; j < 2; ++j)
#pragma unroll
      for (int r = 0; r < 8; ++r) {
        const int qi = sm * 16 + ((lane < 16) ? r : r + 8);
        const int ki = (sn0 + j) * 16 + (lane & 15);
        float s = accs[j][r] * scale;
        if (ki > qi) s = -1.0e9f;
        sc[qi][ki] = s;
      }
    __syncthreads();

    // softmax rows (fp32) -> pb (bf16)
    if (tid < SEQ) {
      float mx = sc[tid][0];
      for (int c = 1; c < SEQ; ++c) mx = fmaxf(mx, sc[tid][c]);
      float sum = 0.f;
      for (int c = 0; c < SEQ; ++c) { float e = __expf(sc[tid][c] - mx); sc[tid][c] = e; sum += e; }
      float inv = 1.f / sum;
      for (int c = 0; c < SEQ; ++c) pb[tid][c] = f2bf(sc[tid][c] * inv);
    }
    __syncthreads();

    // O = P @ V
    const int pm  = wave >> 1;
    const int pn0 = (wave & 1) * 4;      // 4 d-subtiles of 16
    v8f acco[4] = { v8f_zero(), v8f_zero(), v8f_zero(), v8f_zero() };
    const int prow = pm * 16 + (lane & 15);
#pragma unroll
    for (int kb = 0; kb < SEQ; kb += 32) {
      v16bf af = load_a_frag(&pb[prow][kb], lane);
#pragma unroll
      for (int j = 0; j < 4; ++j) {
        const int dcol = (pn0 + j) * 16 + (lane & 15);
        v16bf bfrag = load_b_frag(&vt[dcol][kb], lane);
        acco[j] = wmma_bf16(af, bfrag, acco[j]);
      }
    }
#pragma unroll
    for (int j = 0; j < 4; ++j)
#pragma unroll
      for (int r = 0; r < 8; ++r) {
        const int qi = pm * 16 + ((lane < 16) ? r : r + 8);
        const int dc = (pn0 + j) * 16 + (lane & 15);
        o[(size_t)(tok0 + qi) * (N_HEADS * HEAD_DIM) + head * HEAD_DIM + dc] = f2bf(acco[j][r]);
      }
    __syncthreads();
  }
}

// ---------------- fp32 output head + de-normalization + transpose ----------------
__global__ void head_kernel(const bf16* __restrict__ hf, const float* __restrict__ out_w,
                            const float* __restrict__ out_b, const float* __restrict__ stats,
                            float* __restrict__ out) {
  const int p = blockIdx.x;   // 0..95
  const int r = blockIdx.y;   // 0..27 = b*7+m
  __shared__ float red[256];
  const bf16* row = hf + (size_t)r * OUTK;
  float acc = 0.f;
  for (int j = threadIdx.x; j < OUTK; j += 256)
    acc += bf2f(row[j]) * out_w[(size_t)j * PRED + p];
  red[threadIdx.x] = acc; __syncthreads();
  for (int s = 128; s > 0; s >>= 1) {
    if (threadIdx.x < s) red[threadIdx.x] += red[threadIdx.x + s];
    __syncthreads();
  }
  if (threadIdx.x == 0) {
    float v = red[0] + out_b[p];
    const int b = r / 7, m = r % 7;
    out[((size_t)b * PRED + p) * 7 + m] = v * stats[r * 2 + 1] + stats[r * 2];
  }
}

extern "C" void kernel_launch(void* const* d_in, const int* in_sizes, int n_in,
                              void* d_out, int out_size, void* d_ws, size_t ws_size,
                              hipStream_t stream) {
  (void)in_sizes; (void)n_in; (void)out_size; (void)ws_size;
  const float* x        = (const float*)d_in[0];
  const float* in_w     = (const float*)d_in[2];
  const float* in_b     = (const float*)d_in[3];
  const float* out_w    = (const float*)d_in[4];
  const float* out_b    = (const float*)d_in[5];
  const bf16*  final_ln = (const bf16*)d_in[6];

  char* ws = (char*)d_ws;
  size_t off = 0;
  auto alloc = [&](size_t bytes) -> char* {
    char* p = ws + off;
    off = (off + bytes + 255) & ~(size_t)255;
    return p;
  };
  float* stats = (float*)alloc(NSER * 2 * sizeof(float));
  bf16* h  = (bf16*)alloc((size_t)NTOK * D_MODEL * 2);
  bf16* hn = (bf16*)alloc((size_t)NTOK * D_MODEL * 2);
  bf16* qb = (bf16*)alloc((size_t)NTOK * D_MODEL * 2);
  bf16* kb = (bf16*)alloc((size_t)NTOK * 512 * 2);
  bf16* vb = (bf16*)alloc((size_t)NTOK * 512 * 2);
  bf16* ao = (bf16*)alloc((size_t)NTOK * D_MODEL * 2);
  bf16* ff = (bf16*)alloc((size_t)NTOK * D_FF * 2);

  stats_kernel<<<NSER, 256, 0, stream>>>(x, stats);
  embed_kernel<<<NTOK, 256, 0, stream>>>(x, stats, in_w, in_b, h);

  for (int l = 0; l < 2; ++l) {
    const int base = 7 + l * 12;
    const bf16* ln1    = (const bf16*)d_in[base + 0];
    const bf16* q_w    = (const bf16*)d_in[base + 1];
    const bf16* q_bias = (const bf16*)d_in[base + 2];
    const bf16* k_w    = (const bf16*)d_in[base + 3];
    const bf16* k_bias = (const bf16*)d_in[base + 4];
    const bf16* v_w    = (const bf16*)d_in[base + 5];
    const bf16* v_bias = (const bf16*)d_in[base + 6];
    const bf16* o_w    = (const bf16*)d_in[base + 7];
    const bf16* ln2    = (const bf16*)d_in[base + 8];
    const bf16* gate_w = (const bf16*)d_in[base + 9];
    const bf16* up_w   = (const bf16*)d_in[base + 10];
    const bf16* down_w = (const bf16*)d_in[base + 11];

    rmsnorm_kernel<<<NTOK, 256, 0, stream>>>(h, ln1, hn);
    gemm_bf16<0><<<dim3(D_MODEL / 128, NTOK / 128), 256, 0, stream>>>(hn, q_w, qb, q_bias, NTOK, D_MODEL, D_MODEL);
    gemm_bf16<0><<<dim3(512 / 128,     NTOK / 128), 256, 0, stream>>>(hn, k_w, kb, k_bias, NTOK, 512, D_MODEL);
    gemm_bf16<0><<<dim3(512 / 128,     NTOK / 128), 256, 0, stream>>>(hn, v_w, vb, v_bias, NTOK, 512, D_MODEL);
    rope_kernel<<<(NTOK * N_HEADS * 64 + 255) / 256, 256, 0, stream>>>(qb, N_HEADS, NTOK);
    rope_kernel<<<(NTOK * N_KV    * 64 + 255) / 256, 256, 0, stream>>>(kb, N_KV, NTOK);
    attn_kernel<<<NSER * N_KV, 256, 0, stream>>>(qb, kb, vb, ao);
    gemm_bf16<1><<<dim3(D_MODEL / 128, NTOK / 128), 256, 0, stream>>>(ao, o_w, h, nullptr, NTOK, D_MODEL, D_MODEL);
    rmsnorm_kernel<<<NTOK, 256, 0, stream>>>(h, ln2, hn);
    gemm_bf16<0><<<dim3(D_FF / 128, NTOK / 128), 256, 0, stream>>>(hn, gate_w, ff, nullptr, NTOK, D_FF, D_MODEL);
    gemm_bf16<2><<<dim3(D_FF / 128, NTOK / 128), 256, 0, stream>>>(hn, up_w, ff, nullptr, NTOK, D_FF, D_MODEL);
    gemm_bf16<1><<<dim3(D_MODEL / 128, NTOK / 128), 256, 0, stream>>>(ff, down_w, h, nullptr, NTOK, D_MODEL, D_FF);
  }

  rmsnorm_kernel<<<NTOK, 256, 0, stream>>>(h, final_ln, hn);
  head_kernel<<<dim3(PRED, NSER), 256, 0, stream>>>(hn, out_w, out_b, stats, (float*)d_out);
}
